// CapsNetBasic_9955734192207
// MI455X (gfx1250) — matrix operand order repaired and checked by reading, not compiled, not run on hardware
//
#include <hip/hip_runtime.h>

typedef __attribute__((ext_vector_type(16))) _Float16 v16h;
typedef __attribute__((ext_vector_type(8)))  _Float16 v8h;
typedef __attribute__((ext_vector_type(8)))  float    v8f;

#define HWD 128
#define PW  132             // padded width/height (2-halo each side)
#define NPIX 32768          // B*H*W = 2*128*128
#define ACT_HALVES (2 * PW * PW * 256)   // 8,921,088 f16 in padded activation buffer
#define SROW 264            // LDS row stride in halves (256 + 8 pad -> conflict-free)
#define SBUF (32 * SROW)    // halves per LDS stage buffer

// ---------------- K0: repack Wp [256oc][256ic][5][5] f32 -> Wt [25][256oc][256ic'] f16,
// where within each 32-wide ic chunk the order is [0..7, 16..23, 8..15, 24..31] so each
// lane's 16-half A-fragment (ISA 7.12.2 16-bit A layout) is one contiguous 32B load.
__global__ void k_prep_wt(const float* __restrict__ wp, _Float16* __restrict__ wt) {
  int idx = blockIdx.x * 256 + threadIdx.x;       // 25*256*256 = 1,638,400
  if (idx >= 25 * 256 * 256) return;
  int icp = idx & 255, oc = (idx >> 8) & 255, kk = idx >> 16;
  int pos = icp & 31, chunk = icp & ~31;
  int g = pos >> 3;
  int gsrc = (g == 1) ? 2 : (g == 2) ? 1 : g;     // swap middle two 8-groups
  int ic = chunk + gsrc * 8 + (pos & 7);
  wt[idx] = (_Float16)wp[(oc * 256 + ic) * 25 + kk];
}

// ---------------- Kz: zero the padded activation buffer (halo must read as 0)
__global__ void k_zero_act(_Float16* __restrict__ actH) {
  int gid = blockIdx.x * 256 + threadIdx.x;       // ACT_HALVES/8 = 1,115,136 threads
  v8h z;
#pragma unroll
  for (int i = 0; i < 8; ++i) z[i] = (_Float16)0.f;
  *(v8h*)(actH + (size_t)gid * 8) = z;
}

// ---------------- K1: conv1 5x5 pad2 (1->256) + bias + ReLU -> padded NHWC f16
// actH[b][y+2][x+2][256]
__global__ void k_conv1(const float* __restrict__ xin, const float* __restrict__ w1,
                        const float* __restrict__ b1, _Float16* __restrict__ actH) {
  int gid = blockIdx.x * 256 + threadIdx.x;       // 32768 pixels * 32 channel-groups
  int p  = gid >> 5;
  int cg = gid & 31;                              // group of 8 output channels
  int b = p >> 14, y = (p >> 7) & 127, x = p & 127;
  float patch[25];
#pragma unroll
  for (int ky = 0; ky < 5; ++ky)
#pragma unroll
    for (int kx = 0; kx < 5; ++kx) {
      int yy = y + ky - 2, xx = x + kx - 2;
      bool ok = (yy >= 0) & (yy < HWD) & (xx >= 0) & (xx < HWD);
      patch[ky * 5 + kx] = ok ? xin[(b << 14) + (yy << 7) + xx] : 0.f;
    }
  v8h h;
#pragma unroll
  for (int j = 0; j < 8; ++j) {
    int co = cg * 8 + j;
    float s = b1[co];
    const float* w = w1 + co * 25;
#pragma unroll
    for (int k = 0; k < 25; ++k) s += w[k] * patch[k];
    h[j] = (_Float16)fmaxf(s, 0.f);
  }
  size_t ppad = (size_t)(b * PW + y + 2) * PW + (x + 2);
  *(v8h*)(actH + ppad * 256 + cg * 8) = h;        // 16B aligned store
}

// squash-and-store one 16x16 C tile's lane-half capsule (8 atoms in the 8 C-VGPRs)
__device__ __forceinline__ void squash_store(const v8f& acc, const float* cb, float* o) {
  const float inv32 = 1.f / 32.f;
  float pre[8];
  float sq = 0.f;
#pragma unroll
  for (int r = 0; r < 8; ++r) { pre[r] = acc[r] * inv32 + cb[r]; sq += pre[r] * pre[r]; }
  float sc = (sq / (1.f + sq)) * rsqrtf(sq + 1e-9f);
  float4 s;
  s.x = pre[0]*sc; s.y = pre[1]*sc; s.z = pre[2]*sc; s.w = pre[3]*sc; *(float4*)o = s;
  s.x = pre[4]*sc; s.y = pre[5]*sc; s.z = pre[6]*sc; s.w = pre[7]*sc; *(float4*)(o+4) = s;
}

// load one 32B B-fragment from LDS as two 16B halves (16B row alignment, bank-tiled)
__device__ __forceinline__ v16h lds_frag(const _Float16* p) {
  v8h lo = *(const v8h*)(p);
  v8h hi = *(const v8h*)(p + 16);
  return __builtin_shufflevector(lo, hi, 0, 1, 2, 3, 4, 5, 6, 7,
                                 8, 9, 10, 11, 12, 13, 14, 15);
}

// ---------------- K2: primary capsule conv as 25 shifted GEMMs via WMMA f16->f32.
// Block: 256 thr = 8 waves; wave w -> 32 oc x 32 px tile (2 A-frags x 2 B-frags = 4 WMMA
// per chunk). B (shared by all 8 waves) is staged through a double-buffered LDS tile of
// one kernel-tap (32 cols x 256 ic f16, rows padded to 264 halves -> conflict-free
// ds_load_b128). Producer global loads for tap kk+1 stay in flight across the whole
// 32-WMMA consumption of tap kk; B waits hit DScnt only, decoupled from the A-prefetch
// LOADcnt pipeline. A fragments load direct from the L2-resident permuted weights.
__global__ void k_primary_wmma(const _Float16* __restrict__ actH,
                               const _Float16* __restrict__ wt,
                               const float* __restrict__ bp,
                               const float* __restrict__ cbp,
                               float* __restrict__ prim) {
  __shared__ _Float16 sB[2 * SBUF];               // 2 x 16.5KB
  int tid  = threadIdx.x;
  int lane = tid & 31;
  int wv   = tid >> 5;
  int lj = lane & 15, hi = lane >> 4;
  int oc_base = wv * 32;                          // 8 waves cover all 256 oc
  int n_base  = blockIdx.x * 32;

  // ---- producer mapping: thread -> (column c = tid>>3, ic chunk = (tid&7)*32)
  {
  }
  int pc  = n_base + (tid >> 3);
  int pcb = pc >> 14, pcy = (pc >> 7) & 127, pcx = pc & 127;
  const _Float16* gcol =
      actH + ((size_t)((pcb * PW + pcy + 2) * PW + (pcx + 2)) * 256 + (tid & 7) * 32);
  _Float16* scol = &sB[(tid >> 3) * SROW + (tid & 7) * 32];

  // ---- consumer mapping: per-lane pixel column of the two 16-col tiles
  int p0 = n_base + lj;
  int p1 = n_base + 16 + lj;
  const _Float16* c0 = &sB[lj * SROW + hi * 8];
  const _Float16* c1 = &sB[(16 + lj) * SROW + hi * 8];

  const _Float16* wrow0 = wt + (size_t)(oc_base + lj) * 256 + hi * 16;
  const _Float16* wrow1 = wt + (size_t)(oc_base + 16 + lj) * 256 + hi * 16;

  v8f acc00, acc01, acc10, acc11;
#pragma unroll
  for (int r = 0; r < 8; ++r) { acc00[r] = 0.f; acc01[r] = 0.f; acc10[r] = 0.f; acc11[r] = 0.f; }

  // preload tap kk=0 into buffer 0
  {
    const int koff = (-2 * PW - 2) * 256;
    v8h g0 = *(const v8h*)(gcol + koff);
    v8h g1 = *(const v8h*)(gcol + koff + 8);
    v8h g2 = *(const v8h*)(gcol + koff + 16);
    v8h g3 = *(const v8h*)(gcol + koff + 24);
    *(v8h*)(scol)      = g0;
    *(v8h*)(scol + 8)  = g1;
    *(v8h*)(scol + 16) = g2;
    *(v8h*)(scol + 24) = g3;
  }
  __syncthreads();

#pragma unroll 1
  for (int kk = 0; kk < 25; ++kk) {
    int cur = kk & 1;
    // issue producer global loads for tap kk+1 (in flight during consumption below)
    v8h g0, g1, g2, g3;
    bool more = (kk + 1 < 25);
    if (more) {
      int k1 = kk + 1;
      int dy = k1 / 5 - 2, dx = k1 % 5 - 2;
      int koff = (dy * PW + dx) * 256;
      g0 = *(const v8h*)(gcol + koff);
      g1 = *(const v8h*)(gcol + koff + 8);
      g2 = *(const v8h*)(gcol + koff + 16);
      g3 = *(const v8h*)(gcol + koff + 24);
    }
    // consume tap kk from LDS: 8 ic-chunks x 4 WMMA
    const _Float16* bc0 = c0 + cur * SBUF;
    const _Float16* bc1 = c1 + cur * SBUF;
    const _Float16* pA0 = wrow0 + (size_t)kk * 65536;
    const _Float16* pA1 = wrow1 + (size_t)kk * 65536;
#pragma unroll
    for (int ic0 = 0; ic0 < 256; ic0 += 32) {
      v16h a0 = *(const v16h*)(pA0 + ic0);
      v16h a1 = *(const v16h*)(pA1 + ic0);
      v16h b0 = lds_frag(bc0 + ic0);
      v16h b1 = lds_frag(bc1 + ic0);
      acc00 = __builtin_amdgcn_wmma_f32_16x16x32_f16(false, a0, false, b0, (short)0, acc00,
                                                     false, false);
      acc01 = __builtin_amdgcn_wmma_f32_16x16x32_f16(false, a0, false, b1, (short)0, acc01,
                                                     false, false);
      acc10 = __builtin_amdgcn_wmma_f32_16x16x32_f16(false, a1, false, b0, (short)0, acc10,
                                                     false, false);
      acc11 = __builtin_amdgcn_wmma_f32_16x16x32_f16(false, a1, false, b1, (short)0, acc11,
                                                     false, false);
    }
    // commit tap kk+1 into the back buffer, then block-wide handoff
    if (more) {
      _Float16* sc = scol + (cur ^ 1) * SBUF;
      *(v8h*)(sc)      = g0;
      *(v8h*)(sc + 8)  = g1;
      *(v8h*)(sc + 16) = g2;
      *(v8h*)(sc + 24) = g3;
    }
    __syncthreads();
  }

  // epilogue: lane-half holds the 8 atoms of one capsule per oc-tile
  const float inv32 = 1.f / 32.f;
  int rowA = oc_base + hi * 8;          // oc-tile 0
  int rowB = oc_base + 16 + hi * 8;     // oc-tile 1
  float cbA[8], cbB[8];
#pragma unroll
  for (int r = 0; r < 8; ++r) {
    cbA[r] = bp[rowA + r] * inv32 + cbp[rowA + r];
    cbB[r] = bp[rowB + r] * inv32 + cbp[rowB + r];
  }
  squash_store(acc00, cbA, prim + (size_t)p0 * 256 + rowA);
  squash_store(acc01, cbA, prim + (size_t)p1 * 256 + rowA);
  squash_store(acc10, cbB, prim + (size_t)p0 * 256 + rowB);
  squash_store(acc11, cbB, prim + (size_t)p1 * 256 + rowB);
}

// ---------------- K3: fused tail. Per pixel:
//   seg preact[a'] = sum_{oc} Ws[a'][oc&7]*prim[oc] + 32*bs + cbs  (routing over out_dim=1
//   is identity: softmax over a singleton axis == 1 for every iteration)
//   -> squash16 -> ||.|| = out_seg ; mask by y ; 16->64 relu ; 64->128 relu ; 128->1 sigmoid
__global__ void k_tail(const float* __restrict__ prim, const float* __restrict__ yin,
                       const float* __restrict__ Ws,  const float* __restrict__ bs,
                       const float* __restrict__ cbs,
                       const float* __restrict__ Wr1, const float* __restrict__ br1,
                       const float* __restrict__ Wr2, const float* __restrict__ br2,
                       const float* __restrict__ Wr3, const float* __restrict__ br3,
                       float* __restrict__ out) {
  __shared__ float sWs[128], sbs[16], scbs[16];
  __shared__ float sWr1[1024], sbr1[64];
  __shared__ float sWr2[8192], sbr2[128];
  __shared__ float sWr3[128];
  int tid = threadIdx.x;
  for (int i = tid; i < 8192; i += 256) sWr2[i] = Wr2[i];
  for (int i = tid; i < 1024; i += 256) sWr1[i] = Wr1[i];
  if (tid < 128) { sWs[tid] = Ws[tid]; sWr3[tid] = Wr3[tid]; sbr2[tid] = br2[tid]; }
  if (tid < 64)  sbr1[tid] = br1[tid];
  if (tid < 16)  { sbs[tid] = bs[tid]; scbs[tid] = cbs[tid]; }
  __syncthreads();

  int p = blockIdx.x * 256 + tid;
  const float4* pp = (const float4*)(prim + (size_t)p * 256);

  float acc[16];
#pragma unroll
  for (int a = 0; a < 16; ++a) acc[a] = 32.f * sbs[a] + scbs[a];
  for (int i = 0; i < 32; ++i) {                  // 32 input capsules x 8 atoms
    float4 q0 = pp[i * 2];
    float4 q1 = pp[i * 2 + 1];
#pragma unroll
    for (int a = 0; a < 16; ++a) {
      const float* w = sWs + a * 8;
      acc[a] += w[0]*q0.x + w[1]*q0.y + w[2]*q0.z + w[3]*q0.w
              + w[4]*q1.x + w[5]*q1.y + w[6]*q1.z + w[7]*q1.w;
    }
  }
  float sq = 0.f;
#pragma unroll
  for (int a = 0; a < 16; ++a) sq += acc[a] * acc[a];
  float sc = (sq / (1.f + sq)) * rsqrtf(sq + 1e-9f);
  out[p] = sqrtf(sq * sc * sc + 1e-9f);           // ||squash(preact)||

  float yv = yin[p];
  float mk[16];
#pragma unroll
  for (int a = 0; a < 16; ++a) mk[a] = acc[a] * sc * yv;

  float r1v[64];
#pragma unroll
  for (int j = 0; j < 64; ++j) {
    float t = sbr1[j];
    const float* w = sWr1 + j * 16;
#pragma unroll
    for (int a = 0; a < 16; ++a) t += w[a] * mk[a];
    r1v[j] = fmaxf(t, 0.f);
  }
  float accR = br3[0];
  for (int c = 0; c < 128; ++c) {                 // r2 never materialized
    float t = sbr2[c];
    const float* w = sWr2 + c * 64;
#pragma unroll
    for (int j = 0; j < 64; ++j) t += w[j] * r1v[j];
    accR += sWr3[c] * fmaxf(t, 0.f);
  }
  out[NPIX + p] = 1.f / (1.f + __expf(-accR));
}

extern "C" void kernel_launch(void* const* d_in, const int* in_sizes, int n_in,
                              void* d_out, int out_size, void* d_ws, size_t ws_size,
                              hipStream_t stream) {
  const float* x   = (const float*)d_in[0];
  const float* y   = (const float*)d_in[1];
  const float* W1  = (const float*)d_in[2];
  const float* b1  = (const float*)d_in[3];
  const float* Wp  = (const float*)d_in[4];
  const float* bp  = (const float*)d_in[5];
  const float* cbp = (const float*)d_in[6];
  const float* Ws  = (const float*)d_in[7];
  const float* bs  = (const float*)d_in[8];
  const float* cbs = (const float*)d_in[9];
  const float* Wr1 = (const float*)d_in[10];
  const float* br1 = (const float*)d_in[11];
  const float* Wr2 = (const float*)d_in[12];
  const float* br2 = (const float*)d_in[13];
  const float* Wr3 = (const float*)d_in[14];
  const float* br3 = (const float*)d_in[15];
  float* out = (float*)d_out;

  // workspace layout (needs ~54.7 MB):
  //   [0, 3276800)                  Wt   f16 [25][256][256]  (fragment-permuted)
  //   [3276800, +17842176)          actH f16 [2][132][132][256]  (zero-padded halo NHWC)
  //   [21118976, +33554432)         prim f32 [32768][256]    (squashed primary caps, NHWC)
  char* ws = (char*)d_ws;
  _Float16* wt   = (_Float16*)(ws);
  _Float16* actH = (_Float16*)(ws + 3276800);
  float*    prim = (float*)(ws + 21118976);

  k_prep_wt<<<6400, 256, 0, stream>>>(Wp, wt);
  k_zero_act<<<ACT_HALVES / (8 * 256), 256, 0, stream>>>(actH);
  k_conv1<<<4096, 256, 0, stream>>>(x, W1, b1, actH);
  k_primary_wmma<<<NPIX / 32, 256, 0, stream>>>(actH, wt, bp, cbp, prim);
  k_tail<<<NPIX / 256, 256, 0, stream>>>(prim, y, Ws, bs, cbs, Wr1, br1, Wr2, br2, Wr3, br3, out);
}
